// MultiScaleAttention_16535624089602
// MI455X (gfx1250) — matrix operand order repaired and checked by reading, not compiled, not run on hardware
//
#include <hip/hip_runtime.h>
#include <math.h>

#define NH   8
#define HD   96
#define BB   2
#define TT   8
#define HS   56
#define WS   56
#define L_IN (TT*HS*WS)       /* 25088 */
#define C    768
#define TQ   8
#define HQ   28
#define WQ   28
#define LQ   (TQ*HQ*WQ)       /* 6272  */
#define TK   8
#define HK   7
#define WK   7
#define LK   (TK*HK*WK)       /* 392   */
#define LKP  416              /* 13*32 padded */
#define ATT_SCALE 0.102062072615965f  /* 96^-0.5 */

typedef __attribute__((ext_vector_type(16))) __bf16 v16bf;
typedef __attribute__((ext_vector_type(8)))  float  v8f;
typedef __attribute__((ext_vector_type(8)))  __bf16 bf16x8;

__device__ __forceinline__ __bf16 f2bf(float f) {
  union { float f; unsigned u; } a; a.f = f;
  unsigned u = a.u + (0x7FFFu + ((a.u >> 16) & 1u));   // RNE
  union { unsigned short s; __bf16 b; } o; o.s = (unsigned short)(u >> 16);
  return o.b;
}
__device__ __forceinline__ float bf2f(__bf16 b) {
  union { unsigned short s; __bf16 b; } i; i.b = b;
  union { unsigned u; float f; } o; o.u = ((unsigned)i.s) << 16;
  return o.f;
}

__device__ __forceinline__ v8f wmma_bf16(v16bf a, v16bf b, v8f c) {
  return __builtin_amdgcn_wmma_f32_16x16x32_bf16(false, a, false, b, (short)0, c, false, false);
}

// A fragment 16x32 bf16 from row-major tile (ld elements per row).
// lane<16: row=lane, K {0..7}U{16..23}; lane>=16: row=lane-16, K {8..15}U{24..31}
__device__ __forceinline__ v16bf load_a_bf16(const __bf16* base, int ld, int lane) {
  int row  = lane & 15;
  int koff = (lane & 16) ? 8 : 0;
  const __bf16* p = base + (size_t)row * ld + koff;
  bf16x8 lo = *(const bf16x8*)(p);
  bf16x8 hi = *(const bf16x8*)(p + 16);
  v16bf r;
#pragma unroll
  for (int j = 0; j < 8; ++j) { r[j] = lo[j]; r[j + 8] = hi[j]; }
  return r;
}

// Same A layout, source is f32, convert on the fly.
__device__ __forceinline__ v16bf load_a_f32(const float* base, int ld, int lane) {
  int row  = lane & 15;
  int koff = (lane & 16) ? 8 : 0;
  const float* p = base + (size_t)row * ld + koff;
  v16bf r;
#pragma unroll
  for (int j = 0; j < 8; ++j) { r[j] = f2bf(p[j]); r[j + 8] = f2bf(p[j + 16]); }
  return r;
}

// B fragment 32x16 bf16 from B^T row-major (bt[n][k], ld elements per row).
// lane<16: col=lane, K {0..7}U{16..23}; lane>=16: col=lane-16, K {8..15}U{24..31}
__device__ __forceinline__ v16bf load_b_bt(const __bf16* bt, int ld, int lane) {
  int n    = lane & 15;
  int koff = (lane & 16) ? 8 : 0;
  const __bf16* p = bt + (size_t)n * ld + koff;
  bf16x8 lo = *(const bf16x8*)(p);
  bf16x8 hi = *(const bf16x8*)(p + 16);
  v16bf r;
#pragma unroll
  for (int j = 0; j < 8; ++j) { r[j] = lo[j]; r[j + 8] = hi[j]; }
  return r;
}

// ---------------------------------------------------------------- conversions
__global__ void k_cvt_bf16(const float* __restrict__ in, __bf16* __restrict__ out, int n) {
  int i = blockIdx.x * blockDim.x + threadIdx.x;
  if (i < n) out[i] = f2bf(in[i]);
}

// ---------------------------------------------------------------- QKV GEMM
// x_bf [50176 x 768] @ qkv_w^T (qkv_w is [2304 x 768] == B^T layout) + bias.
// Wave tile: 32(M) x 64(N), 8 f32 accumulators; block = 4 waves = 128(M) x 64(N).
// Output scattered to qkv_bf [(s*B+b)*NH+h][L_IN][HD] bf16 (pre-pool layout).
__global__ void __launch_bounds__(128)
k_qkv_gemm(const __bf16* __restrict__ xb, const __bf16* __restrict__ wb,
           const float* __restrict__ bias, __bf16* __restrict__ qkv) {
  int lane = threadIdx.x & 31;
  int wave = threadIdx.x >> 5;
  int m0 = blockIdx.x * 128 + wave * 32;
  int n0 = blockIdx.y * 64;
  v8f acc[2][4] = {};
  const __bf16* arow0 = xb + (size_t)m0 * C;
  const __bf16* arow1 = xb + (size_t)(m0 + 16) * C;
  for (int kk = 0; kk < C; kk += 32) {
    if (kk + 32 < C) {                         // prefetch next K-step of A
      __builtin_prefetch(arow0 + kk + 32, 0, 1);
      __builtin_prefetch(arow1 + kk + 32, 0, 1);
    }
    v16bf a0 = load_a_bf16(arow0 + kk, C, lane);
    v16bf a1 = load_a_bf16(arow1 + kk, C, lane);
#pragma unroll
    for (int t = 0; t < 4; ++t) {
      v16bf b = load_b_bt(wb + (size_t)(n0 + 16 * t) * C + kk, C, lane);
      acc[0][t] = wmma_bf16(a0, b, acc[0][t]);
      acc[1][t] = wmma_bf16(a1, b, acc[1][t]);
    }
  }
  int colb = lane & 15;
  int rbase = (lane & 16) ? 8 : 0;
#pragma unroll
  for (int g = 0; g < 2; ++g) {
#pragma unroll
    for (int t = 0; t < 4; ++t) {
      int col = n0 + 16 * t + colb;          // 0..2303
      int s = col / C;                       // 0=q 1=k 2=v
      int rem = col - s * C;
      int h = rem / HD, d = rem - h * HD;
      float bv = bias[col];
#pragma unroll
      for (int i = 0; i < 8; ++i) {
        int row = m0 + 16 * g + rbase + i;   // global token index b*L_IN + l
        int bidx = row / L_IN, l = row - bidx * L_IN;
        size_t o = ((((size_t)s * BB + bidx) * NH + h) * L_IN + l) * HD + d;
        qkv[o] = f2bf(acc[g][t][i] + bv);
      }
    }
  }
}

// ---------------------------------------------------------------- GELU + depthwise conv3d pool
// mode 0: q -> f32 [B,NH,LQ,HD]; mode 1: k -> bf16 [B,NH,LKP,HD];
// mode 2: v -> bf16 transposed [B,NH,HD,LKP]
__global__ void k_pool(const __bf16* __restrict__ in, const float* __restrict__ w,
                       int sh, int sw, int To, int Ho, int Wo, int mode,
                       float* __restrict__ outf, __bf16* __restrict__ outb) {
  long long idx = (long long)blockIdx.x * blockDim.x + threadIdx.x;
  long long total = (long long)BB * NH * To * Ho * Wo * HD;
  if (idx >= total) return;
  int d = (int)(idx % HD); long long r = idx / HD;
  int x = (int)(r % Wo); r /= Wo;
  int y = (int)(r % Ho); r /= Ho;
  int t = (int)(r % To); r /= To;
  int h = (int)(r % NH); int b = (int)(r / NH);
  const __bf16* src = in + ((size_t)b * NH + h) * (size_t)L_IN * HD;
  float acc = 0.f;
#pragma unroll
  for (int kd = 0; kd < 3; ++kd) {
    int tt = t - 1 + kd;                     // stride_t == 1 always
    if (tt < 0 || tt >= TT) continue;
#pragma unroll
    for (int kh = 0; kh < 3; ++kh) {
      int yy = y * sh - 1 + kh;
      if (yy < 0 || yy >= HS) continue;
#pragma unroll
      for (int kw = 0; kw < 3; ++kw) {
        int xx = x * sw - 1 + kw;
        if (xx < 0 || xx >= WS) continue;
        float v = bf2f(src[(((size_t)tt * HS + yy) * WS + xx) * HD + d]);
        float g = 0.5f * v * (1.0f + erff(v * 0.70710678118f));   // exact GELU
        acc += g * w[d * 27 + (kd * 3 + kh) * 3 + kw];
      }
    }
  }
  int lo = (t * Ho + y) * Wo + x;
  if (mode == 0)      outf[(((size_t)b * NH + h) * LQ  + lo) * HD + d] = acc;
  else if (mode == 1) outb[(((size_t)b * NH + h) * LKP + lo) * HD + d] = f2bf(acc);
  else                outb[(((size_t)b * NH + h) * HD  + d) * LKP + lo] = f2bf(acc);
}

// ---------------------------------------------------------------- fused attention
// One wave handles 16 query rows: scores (26 n-tiles x 3 k-steps WMMA),
// softmax via LDS, P@V (13 k-steps x 6 n-tiles WMMA), + q residual.
__global__ void __launch_bounds__(32)
k_attn(const float* __restrict__ qpool, const __bf16* __restrict__ kpool,
       const __bf16* __restrict__ vt, __bf16* __restrict__ ares) {
  __shared__ __align__(16) float  sS[16 * LKP];
  __shared__ __align__(16) __bf16 sP[16 * LKP];
  int lane = threadIdx.x;
  int mt = blockIdx.x % (LQ / 16);
  int bh = blockIdx.x / (LQ / 16);
  int b = bh / NH, h = bh % NH;
  int m0 = mt * 16;
  const float*  qp = qpool + (size_t)bh * LQ  * HD;
  const __bf16* kp = kpool + (size_t)bh * LKP * HD;
  const __bf16* vp = vt    + (size_t)bh * HD  * LKP;
  int colb = lane & 15, rbase = (lane & 16) ? 8 : 0;

  v16bf afr[3];
#pragma unroll
  for (int j = 0; j < 3; ++j) afr[j] = load_a_f32(qp + (size_t)m0 * HD + 32 * j, HD, lane);

  for (int nt = 0; nt < LKP / 16; ++nt) {
    v8f acc = {};
#pragma unroll
    for (int j = 0; j < 3; ++j) {
      v16bf bfrag = load_b_bt(kp + (size_t)(nt * 16) * HD + 32 * j, HD, lane);
      acc = wmma_bf16(afr[j], bfrag, acc);
    }
#pragma unroll
    for (int i = 0; i < 8; ++i)
      sS[(rbase + i) * LKP + nt * 16 + colb] = acc[i] * ATT_SCALE;
  }
  __syncthreads();

  if (lane < 16) {                               // one lane per query row
    float mx = -3.0e38f;
    for (int cidx = 0; cidx < LK; ++cidx) mx = fmaxf(mx, sS[lane * LKP + cidx]);
    float sum = 0.f;
    for (int cidx = 0; cidx < LK; ++cidx) {
      float e = __expf(sS[lane * LKP + cidx] - mx);
      sS[lane * LKP + cidx] = e; sum += e;
    }
    float rn = 1.0f / sum;
    for (int cidx = 0; cidx < LK;  ++cidx) sP[lane * LKP + cidx] = f2bf(sS[lane * LKP + cidx] * rn);
    for (int cidx = LK; cidx < LKP; ++cidx) sP[lane * LKP + cidx] = f2bf(0.f);
  }
  __syncthreads();

  v8f oacc[6] = {};
  for (int kt = 0; kt < LKP / 32; ++kt) {
    v16bf a = load_a_bf16(sP + kt * 32, LKP, lane);
#pragma unroll
    for (int t = 0; t < 6; ++t) {
      v16bf bfrag = load_b_bt(vp + (size_t)(t * 16) * LKP + kt * 32, LKP, lane);
      oacc[t] = wmma_bf16(a, bfrag, oacc[t]);
    }
  }
#pragma unroll
  for (int t = 0; t < 6; ++t) {
    int dcol = t * 16 + colb;
#pragma unroll
    for (int i = 0; i < 8; ++i) {
      int row = m0 + rbase + i;
      float val = oacc[t][i] + qp[(size_t)row * HD + dcol];   // residual_pool
      ares[((size_t)b * LQ + row) * C + h * HD + dcol] = f2bf(val);
    }
  }
}

// ---------------------------------------------------------------- output projection
// Wave tile 32(M) x 64(N), same structure as QKV GEMM.
__global__ void __launch_bounds__(128)
k_proj_gemm(const __bf16* __restrict__ ab, const __bf16* __restrict__ wb,
            const float* __restrict__ bias, float* __restrict__ out) {
  int lane = threadIdx.x & 31;
  int wave = threadIdx.x >> 5;
  int m0 = blockIdx.x * 128 + wave * 32;
  int n0 = blockIdx.y * 64;
  v8f acc[2][4] = {};
  const __bf16* arow0 = ab + (size_t)m0 * C;
  const __bf16* arow1 = ab + (size_t)(m0 + 16) * C;
  for (int kk = 0; kk < C; kk += 32) {
    if (kk + 32 < C) {
      __builtin_prefetch(arow0 + kk + 32, 0, 1);
      __builtin_prefetch(arow1 + kk + 32, 0, 1);
    }
    v16bf a0 = load_a_bf16(arow0 + kk, C, lane);
    v16bf a1 = load_a_bf16(arow1 + kk, C, lane);
#pragma unroll
    for (int t = 0; t < 4; ++t) {
      v16bf b = load_b_bt(wb + (size_t)(n0 + 16 * t) * C + kk, C, lane);
      acc[0][t] = wmma_bf16(a0, b, acc[0][t]);
      acc[1][t] = wmma_bf16(a1, b, acc[1][t]);
    }
  }
  int colb = lane & 15, rbase = (lane & 16) ? 8 : 0;
#pragma unroll
  for (int g = 0; g < 2; ++g) {
#pragma unroll
    for (int t = 0; t < 4; ++t) {
      int col = n0 + 16 * t + colb;
      float bv = bias[col];
#pragma unroll
      for (int i = 0; i < 8; ++i)
        out[(size_t)(m0 + 16 * g + rbase + i) * C + col] = acc[g][t][i] + bv;
    }
  }
}

// ---------------------------------------------------------------- host
extern "C" void kernel_launch(void* const* d_in, const int* in_sizes, int n_in,
                              void* d_out, int out_size, void* d_ws, size_t ws_size,
                              hipStream_t stream) {
  (void)in_sizes; (void)n_in; (void)out_size; (void)ws_size;
  const float* x      = (const float*)d_in[0];
  const float* qkv_w  = (const float*)d_in[1];
  const float* qkv_b  = (const float*)d_in[2];
  const float* proj_w = (const float*)d_in[3];
  const float* proj_b = (const float*)d_in[4];
  const float* pq_w   = (const float*)d_in[5];
  const float* pk_w   = (const float*)d_in[6];
  const float* pv_w   = (const float*)d_in[7];
  float* out = (float*)d_out;

  const size_t M   = (size_t)BB * L_IN;       // 50176
  const size_t MQ  = (size_t)BB * LQ;         // 12544
  char* ws = (char*)d_ws;
  size_t off = 0;
  auto alloc = [&](size_t bytes) { size_t p = off; off = (off + bytes + 255) & ~(size_t)255; return p; };

  __bf16* x_bf   = (__bf16*)(ws + alloc(M * C * 2));                          // 77 MB
  __bf16* qw_bf  = (__bf16*)(ws + alloc((size_t)3 * C * C * 2));              // 3.5 MB
  __bf16* pw_bf  = (__bf16*)(ws + alloc((size_t)C * C * 2));                  // 1.2 MB
  __bf16* qkv_bf = (__bf16*)(ws + alloc((size_t)3 * BB * NH * L_IN * HD * 2));// 231 MB
  float*  qpool  = (float*) (ws + alloc((size_t)BB * NH * LQ * HD * 4));      // 38.5 MB
  size_t kp_bytes = (size_t)BB * NH * LKP * HD * 2;                           // 1.28 MB
  __bf16* kpool  = (__bf16*)(ws + alloc(kp_bytes));
  __bf16* vtp    = (__bf16*)(ws + alloc(kp_bytes));
  __bf16* aresb  = (__bf16*)(ws + alloc(MQ * C * 2));                         // 19.3 MB

  // 1) fp32 -> bf16 conversions
  {
    int n = (int)(M * C);
    k_cvt_bf16<<<(n + 255) / 256, 256, 0, stream>>>(x, x_bf, n);
    n = 3 * C * C;
    k_cvt_bf16<<<(n + 255) / 256, 256, 0, stream>>>(qkv_w, qw_bf, n);
    n = C * C;
    k_cvt_bf16<<<(n + 255) / 256, 256, 0, stream>>>(proj_w, pw_bf, n);
  }

  // 2) fused QKV projection (WMMA), scatter to per-head pre-pool layout
  k_qkv_gemm<<<dim3((unsigned)(M / 128), (3 * C) / 64), 128, 0, stream>>>(x_bf, qw_bf, qkv_b, qkv_bf);

  // 3) zero K/V pads, then GELU + depthwise conv pools
  hipMemsetAsync(kpool, 0, kp_bytes, stream);
  hipMemsetAsync(vtp,   0, kp_bytes, stream);
  {
    const size_t slice = (size_t)BB * NH * L_IN * HD;
    long long tq = (long long)BB * NH * LQ * HD;                 // 9,633,792
    long long tk = (long long)BB * NH * LK * HD;                 //   602,112
    k_pool<<<(unsigned)((tq + 255) / 256), 256, 0, stream>>>(qkv_bf,             pq_w, 2, 2, TQ, HQ, WQ, 0, qpool, (__bf16*)nullptr);
    k_pool<<<(unsigned)((tk + 255) / 256), 256, 0, stream>>>(qkv_bf + slice,     pk_w, 8, 8, TK, HK, WK, 1, (float*)nullptr, kpool);
    k_pool<<<(unsigned)((tk + 255) / 256), 256, 0, stream>>>(qkv_bf + 2 * slice, pv_w, 8, 8, TK, HK, WK, 2, (float*)nullptr, vtp);
  }

  // 4) fused attention + residual (WMMA), output in [B, LQ, C] bf16
  k_attn<<<BB * NH * (LQ / 16), 32, 0, stream>>>(qpool, kpool, vtp, aresb);

  // 5) output projection (WMMA) -> fp32
  k_proj_gemm<<<dim3((unsigned)(MQ / 128), C / 64), 128, 0, stream>>>(aresb, pw_bf, proj_b, out);
}
